// Head_23287312679519
// MI455X (gfx1250) — matrix operand order repaired and checked by reading, not compiled, not run on hardware
//
#include <hip/hip_runtime.h>

typedef __attribute__((ext_vector_type(16))) _Float16 v16h;
typedef __attribute__((ext_vector_type(8)))  _Float16 v8h;
typedef __attribute__((ext_vector_type(8)))  float    v8f;
typedef __attribute__((ext_vector_type(4)))  float    v4f;
typedef __attribute__((ext_vector_type(4)))  unsigned int u32x4;
typedef __attribute__((ext_vector_type(8)))  int      i32x8;
typedef __attribute__((ext_vector_type(4)))  int      i32x4;

#define WMMA_F32_F16(a, b, c) \
  __builtin_amdgcn_wmma_f32_16x16x32_f16(false, (a), false, (b), (short)0, (c), false, false)

static constexpr int B_ = 8, T_ = 2048, E_ = 1024, H_ = 64;
static constexpr int BT_ = B_ * T_;
#define NEG_BIG (-3.0e38f)

// ---------------------------------------------------------------------------
// TDM: issue a 2-D f16 tile load (global -> LDS) via tensor descriptor.
// D# bitfields per CDNA5 ISA 8.3/8.4:
//   group0: [1:0]=count(1), [63:32]=lds_addr, [120:64]=global_addr, [127:126]=type(2)
//   group1: [17:16]=data_size(1 => 2B), [79:48]=tensor_dim0, [111:80]=tensor_dim1,
//           [127:112]=tile_dim0, [143:128]=tile_dim1, [207:160]=tensor_dim0_stride
// This toolchain exposes the 6-arg builtin (extra int32x8 before cpol); we use
// 2-D tensors only, so groups 2/3 and the extra group are zero.
// ---------------------------------------------------------------------------
__device__ __forceinline__ void tdm_load_2d(unsigned lds_off, const _Float16* gptr,
                                            unsigned tile_d0, unsigned tile_d1,
                                            unsigned tens_d0, unsigned tens_d1,
                                            unsigned long long stride0)
{
  unsigned long long ga = (unsigned long long)(uintptr_t)gptr;
  u32x4 g0;
  g0[0] = 1u;                                                 // count=1 (valid, user mode)
  g0[1] = lds_off;                                            // LDS byte address
  g0[2] = (unsigned)ga;                                       // global_addr[31:0]
  g0[3] = ((unsigned)(ga >> 32) & 0x01FFFFFFu) | (2u << 30);  // global_addr[56:32], type=2
  i32x8 g1;
  g1[0] = (int)(1u << 16);                                    // data_size = 2 bytes
  g1[1] = (int)((tens_d0 & 0xFFFFu) << 16);                   // tensor_dim0[15:0]
  g1[2] = (int)(((tens_d0 >> 16) & 0xFFFFu) | ((tens_d1 & 0xFFFFu) << 16));
  g1[3] = (int)(((tens_d1 >> 16) & 0xFFFFu) | ((tile_d0 & 0xFFFFu) << 16));
  g1[4] = (int)(tile_d1 & 0xFFFFu);                           // tile_dim1 (tile_dim2 = 0)
  g1[5] = (int)(unsigned)(stride0 & 0xFFFFFFFFull);           // tensor_dim0_stride[31:0]
  g1[6] = (int)(unsigned)((stride0 >> 32) & 0xFFFFull);       // stride0[47:32]
  g1[7] = 0;
  i32x4 g2 = {0, 0, 0, 0};
  i32x4 g3 = {0, 0, 0, 0};
  i32x8 g4 = {0, 0, 0, 0, 0, 0, 0, 0};
  __builtin_amdgcn_tensor_load_to_lds(g0, g1, g2, g3, g4, 0);
}

// ---------------------------------------------------------------------------
// Kernel 1: q = x@Wq, k = x@Wk, v = x@Wv  (f32 in, f16 out, f32 WMMA accum)
// W tiles staged transposed into LDS as f16 so B-fragments are contiguous.
// q,k stored row-major [BT,H]; v stored transposed [B,H,T] for attention.
// ---------------------------------------------------------------------------
__global__ __launch_bounds__(128) void proj_kernel(
    const float* __restrict__ x,
    const float* __restrict__ Wq, const float* __restrict__ Wk, const float* __restrict__ Wv,
    _Float16* __restrict__ qh, _Float16* __restrict__ kh, _Float16* __restrict__ vT)
{
  __shared__ __align__(32) _Float16 Wt[3][16 * 512];  // 48 KB: W^T slices (16 cols x 512 e)

  const int lane = threadIdx.x & 31;
  const int wave = threadIdx.x >> 5;
  const int m0   = (blockIdx.x * 4 + wave) * 16;  // 16-row tile of flattened [B*T]
  const int r    = lane & 15;
  const int half = lane >> 4;

  #pragma unroll 1
  for (int nt = 0; nt < 4; ++nt) {           // 4 column tiles of H=64
    v8f accq = {}, acck = {}, accv = {};
    #pragma unroll 1
    for (int eh = 0; eh < 2; ++eh) {         // two 512-wide halves of E=1024
      __syncthreads();
      for (int i = threadIdx.x; i < 3 * 16 * 512; i += 128) {
        int mat = i >> 13;
        int rem = i & 8191;
        int col = rem >> 9;
        int e   = rem & 511;
        const float* Ws = (mat == 0) ? Wq : (mat == 1) ? Wk : Wv;
        Wt[mat][col * 512 + e] = (_Float16)Ws[(size_t)(eh * 512 + e) * H_ + nt * 16 + col];
      }
      __syncthreads();

      const float* xrow = x + (size_t)(m0 + r) * E_ + eh * 512;
      #pragma unroll 2
      for (int e0 = 0; e0 < 512; e0 += 32) {
        v4f x0 = *(const v4f*)(xrow + e0 + half * 16);
        v4f x1 = *(const v4f*)(xrow + e0 + half * 16 + 4);
        v4f x2 = *(const v4f*)(xrow + e0 + half * 16 + 8);
        v4f x3 = *(const v4f*)(xrow + e0 + half * 16 + 12);
        v16h a;
        #pragma unroll
        for (int j = 0; j < 4; ++j) {
          a[j]      = (_Float16)x0[j];
          a[4 + j]  = (_Float16)x1[j];
          a[8 + j]  = (_Float16)x2[j];
          a[12 + j] = (_Float16)x3[j];
        }
        v16h bq = *(const v16h*)&Wt[0][r * 512 + e0 + half * 16];
        v16h bk = *(const v16h*)&Wt[1][r * 512 + e0 + half * 16];
        v16h bv = *(const v16h*)&Wt[2][r * 512 + e0 + half * 16];
        accq = WMMA_F32_F16(a, bq, accq);
        acck = WMMA_F32_F16(a, bk, acck);
        accv = WMMA_F32_F16(a, bv, accv);
      }
    }
    #pragma unroll
    for (int p = 0; p < 8; ++p) {
      int row = m0 + p + 8 * half;
      int col = nt * 16 + r;
      qh[(size_t)row * H_ + col] = (_Float16)accq[p];
      kh[(size_t)row * H_ + col] = (_Float16)acck[p];
      int bi = row >> 11, ti = row & 2047;
      vT[((size_t)(bi * H_ + col)) * T_ + ti] = (_Float16)accv[p];
    }
  }
}

// ---------------------------------------------------------------------------
// Kernel 2: causal flash attention, one wave per 16-query tile, 2 waves/block.
// K/V tiles double-buffered in LDS via TDM (tensor_load_to_lds + TENSORcnt).
// Scores via WMMA (f32 accum), fp32 online softmax with wave-half shuffle
// reductions, P transposed C->A layout through per-wave LDS, P*V via WMMA.
// Per-wave LDS: kbuf 2x(32x64), vbuf 2x(64x32), pbuf 16x32 = 8704 f16.
// ---------------------------------------------------------------------------
__global__ __launch_bounds__(64) void attn_kernel(
    const _Float16* __restrict__ qh, const _Float16* __restrict__ kh,
    const _Float16* __restrict__ vT, float* __restrict__ out)
{
  __shared__ __align__(32) _Float16 smem[2][8704];

  const int lane = threadIdx.x & 31;
  const int wave = threadIdx.x >> 5;
  const int tile = blockIdx.x * 2 + wave;   // 0..1023
  const int bidx = tile >> 7;               // 128 q-tiles per batch
  const int q0   = (tile & 127) * 16;
  const int r    = lane & 15;
  const int half = lane >> 4;

  _Float16* const wbase = &smem[wave][0];
  const unsigned kb_off[2] = { (unsigned)(uintptr_t)(wbase + 0),
                               (unsigned)(uintptr_t)(wbase + 2048) };
  const unsigned vb_off[2] = { (unsigned)(uintptr_t)(wbase + 4096),
                               (unsigned)(uintptr_t)(wbase + 6144) };
  _Float16* const PB = wbase + 8192;

  const _Float16* kbase = kh + (size_t)bidx * T_ * H_;
  const _Float16* vbase = vT + (size_t)bidx * H_ * T_;

  // Q A-fragments for head-dim chunks [0,32) and [32,64)
  const _Float16* qrow = qh + (size_t)(bidx * T_ + q0 + r) * H_;
  v16h aq[2];
  #pragma unroll
  for (int f = 0; f < 2; ++f) {
    v8h lo = *(const v8h*)(qrow + f * 32 + half * 16);
    v8h hi = *(const v8h*)(qrow + f * 32 + half * 16 + 8);
    #pragma unroll
    for (int j = 0; j < 8; ++j) { aq[f][j] = lo[j]; aq[f][8 + j] = hi[j]; }
  }

  v8f o0 = {}, o1 = {}, o2 = {}, o3 = {};
  float mrow[8], lrow[8];
  #pragma unroll
  for (int p = 0; p < 8; ++p) { mrow[p] = NEG_BIG; lrow[p] = 0.0f; }

  const int send = q0 + 16;

  // Prologue: TDM-load first K tile (32x64 contiguous) and V tile (64 rows x 32)
  tdm_load_2d(kb_off[0], kbase + 0, 2048, 1, 2048, 1, 2048);
  tdm_load_2d(vb_off[0], vbase + 0, 32, 64, 32, 64, 2048);

  int db = 0;
  for (int s0 = 0; s0 < send; s0 += 32) {       // 32 keys per chunk
    // Ensure last iteration's LDS reads drained before TDM may overwrite
    asm volatile("s_wait_dscnt 0" ::: "memory");
    const bool nxt = (s0 + 32) < send;
    if (nxt) {
      tdm_load_2d(kb_off[db ^ 1], kbase + (size_t)(s0 + 32) * H_, 2048, 1, 2048, 1, 2048);
      tdm_load_2d(vb_off[db ^ 1], vbase + (s0 + 32), 32, 64, 32, 64, 2048);
      __builtin_amdgcn_s_wait_tensorcnt(2);     // current tiles landed; next 2 in flight
    } else {
      __builtin_amdgcn_s_wait_tensorcnt(0);
    }
    asm volatile("" ::: "memory");
    const _Float16* KB = wbase + db * 2048;
    const _Float16* VB = wbase + 4096 + db * 2048;

    // ---- scores: two 16x16 tiles, K=64 over head dim, B-frags from LDS ----
    v8f sc0 = {}, sc1 = {};
    #pragma unroll
    for (int c = 0; c < 2; ++c) {
      #pragma unroll
      for (int f = 0; f < 2; ++f) {
        v16h bk = *(const v16h*)&KB[(c * 16 + r) * 64 + f * 32 + half * 16];
        if (c == 0) sc0 = WMMA_F32_F16(aq[f], bk, sc0);
        else        sc1 = WMMA_F32_F16(aq[f], bk, sc1);
      }
    }
    // ---- scale + causal mask ----
    float sA[8], sB[8];
    #pragma unroll
    for (int p = 0; p < 8; ++p) {
      int qi = q0 + p + 8 * half;
      float a0 = sc0[p] * 0.125f;               // 64^-0.5
      float a1 = sc1[p] * 0.125f;
      if (s0 + r > qi)      a0 = NEG_BIG;
      if (s0 + 16 + r > qi) a1 = NEG_BIG;
      sA[p] = a0; sB[p] = a1;
    }
    // ---- online softmax; each row lives in one 16-lane half ----
    #pragma unroll
    for (int p = 0; p < 8; ++p) {
      float mx = fmaxf(sA[p], sB[p]);
      #pragma unroll
      for (int d = 1; d < 16; d <<= 1) mx = fmaxf(mx, __shfl_xor(mx, d, 32));
      float nm    = fmaxf(mrow[p], mx);
      float alpha = __expf(mrow[p] - nm);
      float pa    = __expf(sA[p] - nm);
      float pb    = __expf(sB[p] - nm);
      float rs    = pa + pb;
      #pragma unroll
      for (int d = 1; d < 16; d <<= 1) rs += __shfl_xor(rs, d, 32);
      lrow[p] = lrow[p] * alpha + rs;
      mrow[p] = nm;
      o0[p] *= alpha; o1[p] *= alpha; o2[p] *= alpha; o3[p] *= alpha;
      PB[(p + 8 * half) * 32 + r]      = (_Float16)pa;   // C-layout -> LDS
      PB[(p + 8 * half) * 32 + 16 + r] = (_Float16)pb;
    }
    // cross-lane LDS dependency (invisible to compiler): drain DS counter
    asm volatile("s_wait_dscnt 0" ::: "memory");
    v16h ap;
    {
      v8h lo = *(const v8h*)&PB[r * 32 + half * 16];
      v8h hi = *(const v8h*)&PB[r * 32 + half * 16 + 8];
      #pragma unroll
      for (int j = 0; j < 8; ++j) { ap[j] = lo[j]; ap[8 + j] = hi[j]; }
    }
    asm volatile("" ::: "memory");
    // ---- O += P x V over 4 head-dim column tiles, B-frags from LDS ----
    #pragma unroll
    for (int t = 0; t < 4; ++t) {
      v16h bv = *(const v16h*)&VB[(t * 16 + r) * 32 + half * 16];
      v8f& od = (t == 0) ? o0 : (t == 1) ? o1 : (t == 2) ? o2 : o3;
      od = WMMA_F32_F16(ap, bv, od);
    }
    db ^= 1;
  }

  // ---- finalize: divide by row sums, write [B,T,H] f32 ----
  #pragma unroll
  for (int p = 0; p < 8; ++p) {
    float inv = 1.0f / lrow[p];
    int row = bidx * T_ + q0 + p + 8 * half;
    float* orow = out + (size_t)row * H_;
    orow[0  + r] = o0[p] * inv;
    orow[16 + r] = o1[p] * inv;
    orow[32 + r] = o2[p] * inv;
    orow[48 + r] = o3[p] * inv;
  }
}

// ---------------------------------------------------------------------------
extern "C" void kernel_launch(void* const* d_in, const int* in_sizes, int n_in,
                              void* d_out, int out_size, void* d_ws, size_t ws_size,
                              hipStream_t stream) {
  const float* x  = (const float*)d_in[0];
  const float* Wk = (const float*)d_in[1];
  const float* Wq = (const float*)d_in[2];
  const float* Wv = (const float*)d_in[3];
  float* out = (float*)d_out;

  _Float16* qh = (_Float16*)d_ws;           // [BT, H] f16
  _Float16* kh = qh + (size_t)BT_ * H_;     // [BT, H] f16
  _Float16* vT = kh + (size_t)BT_ * H_;     // [B, H, T] f16 (transposed)

  proj_kernel<<<256, 128, 0, stream>>>(x, Wq, Wk, Wv, qh, kh, vT);
  attn_kernel<<<512, 64, 0, stream>>>(qh, kh, vT, out);
}
